// SRGNN_70437463654917
// MI455X (gfx1250) — compile-verified, matrix operand order checked by reading, not run on hardware
//
#include <hip/hip_runtime.h>
#include <hip/hip_bf16.h>
#include <math.h>

// SR-GNN forward for MI455X (gfx1250). B=512, L=50, D=256, STEP=1.
// All matmuls via v_wmma_f32_16x16x32_bf16; operands in bf16; weight tiles
// double-buffered in LDS via gfx1250 async global->LDS copies (fetch of step
// k+1 overlapped with WMMA of step k, one barrier per K-step).

typedef __attribute__((ext_vector_type(16))) __bf16 v16bf;
typedef __attribute__((ext_vector_type(8)))  __bf16 v8bf;
typedef __attribute__((ext_vector_type(8)))  float  v8f;
typedef int srgnn_v4i __attribute__((vector_size(16)));

#define B_  512
#define L_  50
#define D_  256
#define M_  (B_ * L_)   // 25600 rows

#if defined(__gfx1250__) && __has_builtin(__builtin_amdgcn_global_load_async_to_lds_b128)
#define SRGNN_ASYNC_LDS 1
#endif

#define SRGNN_AS1 __attribute__((address_space(1)))
#define SRGNN_AS3 __attribute__((address_space(3)))

static __device__ __forceinline__ float sigmoidf_(float x) {
  return 1.0f / (1.0f + __expf(-x));
}

static __device__ __forceinline__ v8f wmma_bf16(v16bf a, v16bf b, v8f c) {
  return __builtin_amdgcn_wmma_f32_16x16x32_bf16(false, a, false, b, (short)0, c,
                                                 false, false);
}

// 16B global->LDS copy: async on gfx1250, plain copy otherwise.
static __device__ __forceinline__ void srgnn_cp16(const __bf16* g, __bf16* l) {
#if defined(SRGNN_ASYNC_LDS)
  __builtin_amdgcn_global_load_async_to_lds_b128(
      (SRGNN_AS1 srgnn_v4i*)(void*)(const void*)g,
      (SRGNN_AS3 srgnn_v4i*)(void*)l, 0, 0);
#else
  *(v8bf*)l = *(const v8bf*)g;
#endif
}

static __device__ __forceinline__ void srgnn_wait_async() {
#if defined(SRGNN_ASYNC_LDS)
#if __has_builtin(__builtin_amdgcn_s_wait_asynccnt)
  __builtin_amdgcn_s_wait_asynccnt(0);
#else
  asm volatile("s_wait_asynccnt 0x0" ::: "memory");
#endif
#endif
}

// CDNA5 16-bit A/B fragment: lane holds row (lane&15); K chunks
// [base..base+7] and [base+16..base+23], base = 8*(lane>>4). Two 16B loads.
static __device__ __forceinline__ v16bf frag_bf(const __bf16* __restrict__ rowbase,
                                                int kk, int lane) {
  const __bf16* q = rowbase + kk + ((lane >> 4) << 3);
  v8bf c0 = *(const v8bf*)(q);
  v8bf c1 = *(const v8bf*)(q + 16);
  return __builtin_shufflevector(c0, c1, 0, 1, 2, 3, 4, 5, 6, 7,
                                 8, 9, 10, 11, 12, 13, 14, 15);
}

#define WT_LD 40   // 20 words/row: 5n mod 16 -> conflict-free fragment reads
#define AJ_LD 72   // 36 words/row: 9n mod 16 -> conflict-free fragment reads

// ---------------------------------------------------------------------------
// C[M,N] = X[M,K](bf16) * W[N,K](bf16)^T + bias[N](f32)
// grid=(M/64, N/64), block=128 (4 waves). Double-buffered async weight tiles;
// stage of step k+1 overlaps WMMA of step k; one barrier per K-step.
// ---------------------------------------------------------------------------
template <bool STORE_BF>
__global__ void srgnn_gemm_bf(const __bf16* __restrict__ X, int ldX,
                              const __bf16* __restrict__ W, int ldW,
                              const float* __restrict__ bias,
                              void* __restrict__ Cout, int ldC, int K) {
  __shared__ __bf16 wtile[2][64 * WT_LD];
  const int lane = threadIdx.x & 31;
  const int wave = threadIdx.x >> 5;
  const int rowBase = blockIdx.x * 64 + wave * 16;
  const int colBase = blockIdx.y * 64;

  const __bf16* xrow = X + (size_t)(rowBase + (lane & 15)) * ldX;
  const int chunk0 = threadIdx.x * 2;
  const int srow0 = chunk0 >> 2, spart0 = (chunk0 & 3) * 8;
  const int srow1 = (chunk0 + 1) >> 2, spart1 = ((chunk0 + 1) & 3) * 8;
  const __bf16* wsrc0 = W + (size_t)(colBase + srow0) * ldW + spart0;
  const __bf16* wsrc1 = W + (size_t)(colBase + srow1) * ldW + spart1;
  const int ldst0 = srow0 * WT_LD + spart0;
  const int ldst1 = srow1 * WT_LD + spart1;

  // prologue: stage kk=0 into buffer 0
  srgnn_cp16(wsrc0, &wtile[0][ldst0]);
  srgnn_cp16(wsrc1, &wtile[0][ldst1]);

  v8f acc[4] = {v8f{}, v8f{}, v8f{}, v8f{}};
  int buf = 0;
  for (int kk = 0; kk < K; kk += 32, buf ^= 1) {
    srgnn_wait_async();
    __syncthreads();
    if (kk + 32 < K) {   // stage next step into other buffer (overlapped)
      srgnn_cp16(wsrc0 + kk + 32, &wtile[buf ^ 1][ldst0]);
      srgnn_cp16(wsrc1 + kk + 32, &wtile[buf ^ 1][ldst1]);
      __builtin_prefetch(xrow + kk + 32, 0, 3);
    }
    v16bf a = frag_bf(xrow, kk, lane);
    const __bf16* wb = &wtile[buf][0];
#pragma unroll
    for (int t = 0; t < 4; ++t) {
      v16bf b = frag_bf(wb + (t * 16 + (lane & 15)) * WT_LD, 0, lane);
      acc[t] = wmma_bf16(a, b, acc[t]);
    }
  }

  const int half = lane >> 4;
  const int col  = lane & 15;
#pragma unroll
  for (int t = 0; t < 4; ++t) {
    const int c0 = colBase + t * 16 + col;
    const float bv = bias[c0];
#pragma unroll
    for (int r = 0; r < 8; ++r) {
      const size_t idx = (size_t)(rowBase + r + 8 * half) * ldC + c0;
      if (STORE_BF) ((__bf16*)Cout)[idx] = (__bf16)(acc[t][r] + bv);
      else          ((float*)Cout)[idx]  = acc[t][r] + bv;
    }
  }
}

// ---------------------------------------------------------------------------
// Adjacency einsum: out[b,i,:] = sum_j Aslice[b,i,j]*H[b,j,:] + bias (pad 50->64)
// Dense LDS-tiled GEMM; staging vectorized; store guard specialized per wave.
// grid=(B, D/64), block=128.
// ---------------------------------------------------------------------------
__global__ void srgnn_adj_gemm(const float* __restrict__ Aslice,
                               const __bf16* __restrict__ H,
                               const float* __restrict__ bias,
                               __bf16* __restrict__ out) {
  __shared__ __bf16 As[64 * AJ_LD];   // [i][j]
  __shared__ __bf16 Ht[64 * AJ_LD];   // [n_local][j] (transposed H slice)
  const int tid  = threadIdx.x;
  const int lane = tid & 31;
  const int wave = tid >> 5;
  const int b = blockIdx.x;
  const int nBase = blockIdx.y * 64;

  const float*  arow = Aslice + (size_t)b * (L_ * 2 * L_);
  const __bf16* Hb   = H + (size_t)b * L_ * D_ + nBase;

  { // zero-fill (vectorized)
    v8bf z = {};
#pragma unroll
    for (int i = 0; i < 5; ++i) {
      const int v = tid + i * 128;
      if (v < (64 * AJ_LD) / 8) { ((v8bf*)As)[v] = z; ((v8bf*)Ht)[v] = z; }
    }
  }
  __syncthreads();

  // A: 50x50 valid, row stride 100 f32; float2 loads (1250 units)
#pragma unroll
  for (int it = 0; it < 10; ++it) {
    const int u = tid + it * 128;
    if (u < (L_ * L_ + 1) / 2) {   // 1250
      const int i = u / 25, jj = u - i * 25;
      const float2 v = *(const float2*)(arow + i * (2 * L_) + jj * 2);
      As[i * AJ_LD + jj * 2]     = (__bf16)v.x;
      As[i * AJ_LD + jj * 2 + 1] = (__bf16)v.y;
    }
  }
  // H: 50 rows x 64 cols, v8bf loads (400 units), transposed scatter to LDS
#pragma unroll
  for (int it = 0; it < 4; ++it) {
    const int u = tid + it * 128;
    if (u < L_ * 8) {              // 400
      const int j = u >> 3, n8 = (u & 7) * 8;
      const v8bf v = *(const v8bf*)(Hb + (size_t)j * D_ + n8);
#pragma unroll
      for (int e = 0; e < 8; ++e) Ht[(n8 + e) * AJ_LD + j] = v[e];
    }
  }
  __syncthreads();

  v8f acc[4] = {v8f{}, v8f{}, v8f{}, v8f{}};
#pragma unroll
  for (int kk = 0; kk < 64; kk += 32) {
    v16bf a = frag_bf(As + (wave * 16 + (lane & 15)) * AJ_LD, kk, lane);
#pragma unroll
    for (int t = 0; t < 4; ++t) {
      v16bf bf = frag_bf(Ht + (t * 16 + (lane & 15)) * AJ_LD, kk, lane);
      acc[t] = wmma_bf16(a, bf, acc[t]);
    }
  }

  const int half = lane >> 4;
  const int col  = lane & 15;
  __bf16* orow = out + (size_t)b * L_ * (2 * D_) + nBase + col;
  if (wave < 3) {               // rows 16*wave .. 16*wave+15: all < 50 (uniform)
#pragma unroll
    for (int t = 0; t < 4; ++t) {
      const float bv = bias[nBase + t * 16 + col];
#pragma unroll
      for (int r = 0; r < 8; ++r) {
        const int i = wave * 16 + r + 8 * half;
        orow[(size_t)i * (2 * D_) + t * 16] = (__bf16)(acc[t][r] + bv);
      }
    }
  } else if (half == 0) {       // wave 3: only rows 48,49 valid (r<2, half==0)
#pragma unroll
    for (int t = 0; t < 4; ++t) {
      const float bv = bias[nBase + t * 16 + col];
      orow[(size_t)48 * (2 * D_) + t * 16] = (__bf16)(acc[t][0] + bv);
      orow[(size_t)49 * (2 * D_) + t * 16] = (__bf16)(acc[t][1] + bv);
    }
  }
}

// ---------------------------------------------------------------------------
// Fused GRU: per wave one 16x16 tile of hidden_new; gi (K=512) + gh (K=256)
// in-register (6 accumulators); 48x32 gate-weight tiles double-buffered in LDS.
// grid=(M/64, D/16), block=128.
// ---------------------------------------------------------------------------
__global__ void srgnn_gru_fused(const __bf16* __restrict__ inputs,
                                const __bf16* __restrict__ hiddenb,
                                const float* __restrict__ hidden,
                                const __bf16* __restrict__ w_ih,
                                const float* __restrict__ b_ih,
                                const __bf16* __restrict__ w_hh,
                                const float* __restrict__ b_hh,
                                float* __restrict__ out) {
  __shared__ __bf16 wt[2][48 * WT_LD];
  const int tid  = threadIdx.x;
  const int lane = tid & 31;
  const int wave = tid >> 5;
  const int rowBase = blockIdx.x * 64 + wave * 16;
  const int d = blockIdx.y * 16;

  const __bf16* arow1 = inputs  + (size_t)(rowBase + (lane & 15)) * (2 * D_);
  const __bf16* arow2 = hiddenb + (size_t)(rowBase + (lane & 15)) * D_;

  // staging geometry: 192 chunks, threads 0..95 take 2, rest 0
  const bool stg = (tid < 96);
  const int ch0 = tid * 2, ch1 = tid * 2 + 1;
  const int r0 = ch0 >> 2, p0 = (ch0 & 3) * 8;
  const int r1 = ch1 >> 2, p1 = (ch1 & 3) * 8;
  const int g0 = (r0 >> 4) * D_ + d + (r0 & 15);   // weight row index
  const int g1 = (r1 >> 4) * D_ + d + (r1 & 15);

  v8f gir = {}, gii = {}, gin = {}, ghr = {}, ghi = {}, ghn = {};

  // ---- loop 1: gi over inputs, K = 512 ----
  if (stg) {
    srgnn_cp16(w_ih + (size_t)g0 * (2 * D_) + p0, &wt[0][r0 * WT_LD + p0]);
    srgnn_cp16(w_ih + (size_t)g1 * (2 * D_) + p1, &wt[0][r1 * WT_LD + p1]);
  }
  int buf = 0;
  for (int kk = 0; kk < 2 * D_; kk += 32, buf ^= 1) {
    srgnn_wait_async();
    __syncthreads();
    if (kk + 32 < 2 * D_ && stg) {
      srgnn_cp16(w_ih + (size_t)g0 * (2 * D_) + kk + 32 + p0, &wt[buf ^ 1][r0 * WT_LD + p0]);
      srgnn_cp16(w_ih + (size_t)g1 * (2 * D_) + kk + 32 + p1, &wt[buf ^ 1][r1 * WT_LD + p1]);
    }
    v16bf a = frag_bf(arow1, kk, lane);
    const __bf16* wb = &wt[buf][0];
    gir = wmma_bf16(a, frag_bf(wb + ( 0 + (lane & 15)) * WT_LD, 0, lane), gir);
    gii = wmma_bf16(a, frag_bf(wb + (16 + (lane & 15)) * WT_LD, 0, lane), gii);
    gin = wmma_bf16(a, frag_bf(wb + (32 + (lane & 15)) * WT_LD, 0, lane), gin);
  }

  // ---- loop 2: gh over hidden, K = 256 (loop1 ends on buffer 1 -> stage 0) ----
  if (stg) {
    srgnn_cp16(w_hh + (size_t)g0 * D_ + p0, &wt[0][r0 * WT_LD + p0]);
    srgnn_cp16(w_hh + (size_t)g1 * D_ + p1, &wt[0][r1 * WT_LD + p1]);
  }
  buf = 0;
  for (int kk = 0; kk < D_; kk += 32, buf ^= 1) {
    srgnn_wait_async();
    __syncthreads();
    if (kk + 32 < D_ && stg) {
      srgnn_cp16(w_hh + (size_t)g0 * D_ + kk + 32 + p0, &wt[buf ^ 1][r0 * WT_LD + p0]);
      srgnn_cp16(w_hh + (size_t)g1 * D_ + kk + 32 + p1, &wt[buf ^ 1][r1 * WT_LD + p1]);
    }
    v16bf a = frag_bf(arow2, kk, lane);
    const __bf16* wb = &wt[buf][0];
    ghr = wmma_bf16(a, frag_bf(wb + ( 0 + (lane & 15)) * WT_LD, 0, lane), ghr);
    ghi = wmma_bf16(a, frag_bf(wb + (16 + (lane & 15)) * WT_LD, 0, lane), ghi);
    ghn = wmma_bf16(a, frag_bf(wb + (32 + (lane & 15)) * WT_LD, 0, lane), ghn);
  }

  const int half = lane >> 4;
  const int c0 = d + (lane & 15);
  const float bir = b_ih[c0],          bhr = b_hh[c0];
  const float bii = b_ih[D_ + c0],     bhi = b_hh[D_ + c0];
  const float bin = b_ih[2 * D_ + c0], bhn = b_hh[2 * D_ + c0];
#pragma unroll
  for (int r = 0; r < 8; ++r) {
    const size_t rr = (size_t)(rowBase + r + 8 * half);
    const float h  = hidden[rr * D_ + c0];
    const float rg = sigmoidf_(gir[r] + bir + ghr[r] + bhr);
    const float zg = sigmoidf_(gii[r] + bii + ghi[r] + bhi);
    const float ng = tanhf(gin[r] + bin + rg * (ghn[r] + bhn));
    out[rr * D_ + c0] = (1.0f - zg) * h + zg * ng;
  }
}

// ---------------------------------------------------------------------------
// Glue kernels
// ---------------------------------------------------------------------------
__global__ void srgnn_cvt_bf16(const float* __restrict__ src,
                               __bf16* __restrict__ dst, int n) {
  const int i = blockIdx.x * 256 + threadIdx.x;
  if (i < n) dst[i] = (__bf16)src[i];
}

__global__ void srgnn_gather_emb(const int* __restrict__ items,
                                 const float* __restrict__ emb,
                                 float* __restrict__ hidden,
                                 __bf16* __restrict__ hiddenb) {
  const int r = blockIdx.x;
  const float v = emb[(size_t)items[r] * D_ + threadIdx.x];
  hidden[(size_t)r * D_ + threadIdx.x]  = v;
  hiddenb[(size_t)r * D_ + threadIdx.x] = (__bf16)v;
}

__global__ void srgnn_gather_alias(const int* __restrict__ alias,
                                   const float* __restrict__ hid,
                                   float* __restrict__ seqf,
                                   __bf16* __restrict__ seqb) {
  const int r = blockIdx.x;
  const int b = r / L_;
  const int src = b * L_ + alias[r];
  const float v = hid[(size_t)src * D_ + threadIdx.x];
  seqf[(size_t)r * D_ + threadIdx.x] = v;
  seqb[(size_t)r * D_ + threadIdx.x] = (__bf16)v;
}

__global__ void srgnn_take_ht(const int* __restrict__ seqlen,
                              const float* __restrict__ seqf,
                              float* __restrict__ ht,
                              __bf16* __restrict__ htb) {
  const int b = blockIdx.x;
  const int r = b * L_ + (seqlen[b] - 1);
  const float v = seqf[(size_t)r * D_ + threadIdx.x];
  ht[(size_t)b * D_ + threadIdx.x]  = v;
  htb[(size_t)b * D_ + threadIdx.x] = (__bf16)v;
}

__global__ void srgnn_alpha(const float* __restrict__ q1,
                            const float* __restrict__ q2,
                            const float* __restrict__ W3,
                            float* __restrict__ alpha) {
  const int lane = threadIdx.x & 31;
  const int wave = threadIdx.x >> 5;
  const int r = blockIdx.x * 4 + wave;
  const int b = r / L_;
  float acc = 0.0f;
#pragma unroll
  for (int j = 0; j < D_ / 32; ++j) {
    const int dd = lane + 32 * j;
    acc += sigmoidf_(q1[(size_t)b * D_ + dd] + q2[(size_t)r * D_ + dd]) * W3[dd];
  }
#pragma unroll
  for (int off = 16; off >= 1; off >>= 1) acc += __shfl_xor(acc, off, 32);
  if (lane == 0) alpha[r] = acc;
}

__global__ void srgnn_pool_concat(const float* __restrict__ alpha,
                                  const unsigned char* __restrict__ mask,
                                  const float* __restrict__ seqf,
                                  const float* __restrict__ ht,
                                  __bf16* __restrict__ concatb) {
  const int b = blockIdx.x;
  const int dd = threadIdx.x;
  float acc = 0.0f;
  for (int i = 0; i < L_; ++i) {
    if (mask[b * L_ + i])
      acc += alpha[b * L_ + i] * seqf[((size_t)b * L_ + i) * D_ + dd];
  }
  concatb[(size_t)b * (2 * D_) + dd]      = (__bf16)acc;
  concatb[(size_t)b * (2 * D_) + D_ + dd] = (__bf16)ht[(size_t)b * D_ + dd];
}

// ---------------------------------------------------------------------------
extern "C" void kernel_launch(void* const* d_in, const int* in_sizes, int n_in,
                              void* d_out, int out_size, void* d_ws, size_t ws_size,
                              hipStream_t stream) {
  const float* A      = (const float*)d_in[0];
  const int*   items  = (const int*)d_in[1];
  const int*   alias  = (const int*)d_in[2];
  const unsigned char* mask = (const unsigned char*)d_in[3];
  const int*   seqlen = (const int*)d_in[4];
  const float* emb    = (const float*)d_in[5];
  const float* W_ein  = (const float*)d_in[6];
  const float* b_ein  = (const float*)d_in[7];
  const float* W_eout = (const float*)d_in[8];
  const float* b_eout = (const float*)d_in[9];
  const float* w_ih   = (const float*)d_in[10];
  const float* b_ih   = (const float*)d_in[11];
  const float* w_hh   = (const float*)d_in[12];
  const float* b_hh   = (const float*)d_in[13];
  const float* b_iah  = (const float*)d_in[14];
  const float* b_ioh  = (const float*)d_in[15];
  const float* W1     = (const float*)d_in[16];
  const float* b1     = (const float*)d_in[17];
  const float* W2     = (const float*)d_in[18];
  const float* b2     = (const float*)d_in[19];
  const float* W3     = (const float*)d_in[20];
  const float* Wt     = (const float*)d_in[21];
  const float* bt     = (const float*)d_in[22];

  // ---- workspace carve-up ----
  char* p = (char*)d_ws;
  float* hidden  = (float*)p;  p += (size_t)M_ * D_ * 4;
  float* hnew    = (float*)p;  p += (size_t)M_ * D_ * 4;
  float* seqf    = (float*)p;  p += (size_t)M_ * D_ * 4;
  float* q2v     = (float*)p;  p += (size_t)M_ * D_ * 4;
  float* q1v     = (float*)p;  p += (size_t)B_ * D_ * 4;
  float* htv     = (float*)p;  p += (size_t)B_ * D_ * 4;
  float* alphav  = (float*)p;  p += (size_t)M_ * 4;
  __bf16* hiddenb = (__bf16*)p; p += (size_t)M_ * D_ * 2;
  __bf16* hinb    = (__bf16*)p; p += (size_t)M_ * D_ * 2;
  __bf16* houtb   = (__bf16*)p; p += (size_t)M_ * D_ * 2;
  __bf16* inputsb = (__bf16*)p; p += (size_t)M_ * 2 * D_ * 2;
  __bf16* seqb    = (__bf16*)p; p += (size_t)M_ * D_ * 2;
  __bf16* htb     = (__bf16*)p; p += (size_t)B_ * D_ * 2;
  __bf16* concatb = (__bf16*)p; p += (size_t)B_ * 2 * D_ * 2;
  __bf16* weinb   = (__bf16*)p; p += (size_t)D_ * D_ * 2;
  __bf16* weoutb  = (__bf16*)p; p += (size_t)D_ * D_ * 2;
  __bf16* wihb    = (__bf16*)p; p += (size_t)3 * D_ * 2 * D_ * 2;
  __bf16* whhb    = (__bf16*)p; p += (size_t)3 * D_ * D_ * 2;
  __bf16* w1b     = (__bf16*)p; p += (size_t)D_ * D_ * 2;
  __bf16* w2b     = (__bf16*)p; p += (size_t)D_ * D_ * 2;
  __bf16* wtb     = (__bf16*)p; p += (size_t)D_ * 2 * D_ * 2;

  const dim3 blk(128);
  const int DD = D_ * D_;

  srgnn_cvt_bf16<<<(DD + 255) / 256, 256, 0, stream>>>(W_ein, weinb, DD);
  srgnn_cvt_bf16<<<(DD + 255) / 256, 256, 0, stream>>>(W_eout, weoutb, DD);
  srgnn_cvt_bf16<<<(6 * DD + 255) / 256, 256, 0, stream>>>(w_ih, wihb, 6 * DD);
  srgnn_cvt_bf16<<<(3 * DD + 255) / 256, 256, 0, stream>>>(w_hh, whhb, 3 * DD);
  srgnn_cvt_bf16<<<(DD + 255) / 256, 256, 0, stream>>>(W1, w1b, DD);
  srgnn_cvt_bf16<<<(DD + 255) / 256, 256, 0, stream>>>(W2, w2b, DD);
  srgnn_cvt_bf16<<<(2 * DD + 255) / 256, 256, 0, stream>>>(Wt, wtb, 2 * DD);

  srgnn_gather_emb<<<M_, D_, 0, stream>>>(items, emb, hidden, hiddenb);

  srgnn_gemm_bf<true><<<dim3(M_ / 64, D_ / 64), blk, 0, stream>>>(
      hiddenb, D_, weinb, D_, b_ein, hinb, D_, D_);
  srgnn_gemm_bf<true><<<dim3(M_ / 64, D_ / 64), blk, 0, stream>>>(
      hiddenb, D_, weoutb, D_, b_eout, houtb, D_, D_);

  srgnn_adj_gemm<<<dim3(B_, D_ / 64), blk, 0, stream>>>(A,      hinb,  b_iah, inputsb);
  srgnn_adj_gemm<<<dim3(B_, D_ / 64), blk, 0, stream>>>(A + L_, houtb, b_ioh, inputsb + D_);

  srgnn_gru_fused<<<dim3(M_ / 64, D_ / 16), blk, 0, stream>>>(
      inputsb, hiddenb, hidden, wihb, b_ih, whhb, b_hh, hnew);

  srgnn_gather_alias<<<M_, D_, 0, stream>>>(alias, hnew, seqf, seqb);
  srgnn_take_ht<<<B_, D_, 0, stream>>>(seqlen, seqf, htv, htb);

  srgnn_gemm_bf<false><<<dim3(B_ / 64, D_ / 64), blk, 0, stream>>>(
      htb, D_, w1b, D_, b1, q1v, D_, D_);
  srgnn_gemm_bf<false><<<dim3(M_ / 64, D_ / 64), blk, 0, stream>>>(
      seqb, D_, w2b, D_, b2, q2v, D_, D_);

  srgnn_alpha<<<M_ / 4, blk, 0, stream>>>(q1v, q2v, W3, alphav);
  srgnn_pool_concat<<<B_, D_, 0, stream>>>(alphav, mask, seqf, htv, concatb);

  srgnn_gemm_bf<false><<<dim3(B_ / 64, D_ / 64), blk, 0, stream>>>(
      concatb, 2 * D_, wtb, 2 * D_, bt, (float*)d_out, D_, 2 * D_);
}
